// LearnableEdgeAdding_9783935500488
// MI455X (gfx1250) — matrix operand order repaired and checked by reading, not compiled, not run on hardware
//
#include <hip/hip_runtime.h>
#include <hip/hip_bf16.h>

// ---------------------------------------------------------------------------
// LearnableEdgeAdding for MI455X (gfx1250, wave32, WMMA)
//
//  1) cvt_w1_kernel  : one-time f32->f16 column-major conversion of W1 (160KB)
//  2) mlp_wmma_kernel: persistent workgroups; W1 staged into LDS with
//     GLOBAL_LOAD_ASYNC_TO_LDS_B128 (ASYNCcnt path, no VGPR round-trip);
//     fused gather + (S x 320) @ (320 x 256) f16 WMMA + ReLU + 256x2 layer-2
//     + softmax/gumbel/softmax -> poss[S]   (131 GFLOP on the WMMA pipe)
//  3) histogram + threshold + atomic compaction -> top-K_INSERT edges
//  4) hash-table coalesce(mean) of the 2K undirected edges
//  5) emit x, ei, ew into d_out
// ---------------------------------------------------------------------------

typedef __attribute__((ext_vector_type(16))) _Float16 v16h;
typedef __attribute__((ext_vector_type(8)))  float    v8f;

#define N_NODES  100000
#define D_X      128
#define KE_DIM   64
#define E_EDGES  800000
#define S_EDGES  800000
#define H_DIM    256
#define IN_DIM   320            // KE + 2*D
#define K_INSERT 40000
#define ETOT     (E_EDGES + 2*K_INSERT)   // 880000
#define HASH_CAP (1 << 18)
#define NBINS    1024
#define MLP_BLOCKS 1250         // persistent: 5000 waves x 10 tiles = 50000 tiles

// -------------------------- small device helpers ---------------------------

__device__ __forceinline__ float rng01(unsigned s) {
    // stateless hash -> uniform (1e-10, 1): stand-in for threefry gumbel noise
    s ^= s >> 17; s *= 0xed5ad4bbu;
    s ^= s >> 11; s *= 0xac4c1b51u;
    s ^= s >> 15; s *= 0x31848babu;
    s ^= s >> 14;
    return (float)(s >> 8) * (1.0f / 16777216.0f) + 1e-10f;
}

// ------------------ 1) W1 f32 -> f16 column-major (one time) ---------------

__global__ void cvt_w1_kernel(const float* __restrict__ W1, _Float16* __restrict__ w1h)
{
    int g = blockIdx.x * blockDim.x + threadIdx.x;
    if (g < IN_DIM * H_DIM) {
        int k = g / H_DIM;
        int n = g % H_DIM;
        w1h[(size_t)n * IN_DIM + k] = (_Float16)W1[g];
    }
}

// -------------------- 2) fused gather + WMMA MLP kernel --------------------
// block = 128 threads = 4 waves; each wave owns a sequence of 16-edge M-tiles.
// LDS: W1 f16 column-major w1s[n*IN_DIM + k] (160 KB), staged via async-LDS.

__global__ __launch_bounds__(128)
void mlp_wmma_kernel(const float*    __restrict__ x,
                     const float*    __restrict__ eig,
                     const _Float16* __restrict__ w1h,
                     const float*    __restrict__ b1,
                     const float*    __restrict__ W2,
                     const float*    __restrict__ b2,
                     const int*      __restrict__ se,
                     float*          __restrict__ poss_out)
{
    extern __shared__ _Float16 w1s[];   // [H_DIM][IN_DIM] column-major

    // ---- stage W1h into LDS with the CDNA5 async global->LDS engine -------
    // GVS mode: global address is the SGPR base plus the VGPR offset; LDS address comes from the VDST VGPR
    {
        const unsigned lds_base = (unsigned)__builtin_amdgcn_groupstaticsize();
        const unsigned long long gbase = (unsigned long long)(uintptr_t)w1h;
        const int nchunks = (IN_DIM * H_DIM * (int)sizeof(_Float16)) / 16; // 10240
        for (int c = threadIdx.x; c < nchunks; c += blockDim.x) {
            unsigned lo = lds_base + (unsigned)c * 16u;
            unsigned go = (unsigned)c * 16u;
            asm volatile("global_load_async_to_lds_b128 %0, %1, %2"
                         :
                         : "v"(lo), "v"(go), "s"(gbase)
                         : "memory");
        }
        asm volatile("s_wait_asynccnt 0x0" ::: "memory");
    }
    __syncthreads();

    const int lane = threadIdx.x & 31;
    const int wave = threadIdx.x >> 5;
    const int m    = lane & 15;                 // M row owned by this lane
    const int hi   = lane >> 4;                 // half-wave selector
    const int ncol = lane & 15;
    const int nTiles = S_EDGES / 16;            // 50000
    const int tstride = gridDim.x * 4;

    for (int tile = blockIdx.x * 4 + wave; tile < nTiles; tile += tstride) {
        const int er = tile * 16 + m;           // edge index for A-row build
        const int u  = se[er];
        const int v  = se[S_EDGES + er];

        __builtin_prefetch(&x[(size_t)u * D_X], 0, 1);   // global_prefetch_b8
        __builtin_prefetch(&x[(size_t)v * D_X], 0, 1);

        // Build all 10 A fragments (16x32 f16), contiguous K layout:
        //   halves t=0..15 -> K = kk*32 + hi*16 + t
        v16h afrag[10];
#pragma unroll
        for (int kk = 0; kk < 10; ++kk) {
#pragma unroll
            for (int t = 0; t < 16; ++t) {
                const int kg = kk * 32 + hi * 16 + t;
                float val;
                if (kg < KE_DIM) {
                    float d = eig[(size_t)u * KE_DIM + kg] - eig[(size_t)v * KE_DIM + kg];
                    val = d * d;
                } else if (kg < KE_DIM + D_X) {
                    val = x[(size_t)u * D_X + (kg - KE_DIM)];
                } else {
                    val = x[(size_t)v * D_X + (kg - KE_DIM - D_X)];
                }
                afrag[kk][t] = (_Float16)val;
            }
        }

        // layer-2 partial logits, 8 M rows per lane (M = r + hi*8)
        float2 plog[8];
#pragma unroll
        for (int r = 0; r < 8; ++r) plog[r] = make_float2(0.f, 0.f);

        for (int nn = 0; nn < 16; ++nn) {
            v8f acc = {};
            const int ng = nn * 16 + ncol;
            // B fragment: lane holds column N=ng, halves t -> K = kk*32+hi*16+t
            const _Float16* bp = &w1s[(size_t)ng * IN_DIM];
#pragma unroll
            for (int kk = 0; kk < 10; ++kk) {
                v16h bfrag = *(const v16h*)(bp + kk * 32 + hi * 16);
                acc = __builtin_amdgcn_wmma_f32_16x16x32_f16(
                          false, afrag[kk], false, bfrag,
                          (short)0, acc, false, false);
            }
            const float bb  = b1[ng];
            const float w20 = W2[ng * 2 + 0];
            const float w21 = W2[ng * 2 + 1];
#pragma unroll
            for (int r = 0; r < 8; ++r) {
                float h = fmaxf(acc[r] + bb, 0.f);      // ReLU
                plog[r].x = fmaf(h, w20, plog[r].x);
                plog[r].y = fmaf(h, w21, plog[r].y);
            }
        }

        // butterfly-reduce over the 16 lanes sharing each (hi, r) row
#pragma unroll
        for (int mask = 1; mask < 16; mask <<= 1) {
#pragma unroll
            for (int r = 0; r < 8; ++r) {
                plog[r].x += __shfl_xor(plog[r].x, mask, 32);
                plog[r].y += __shfl_xor(plog[r].y, mask, 32);
            }
        }

        if (ncol == 0) {
            const float bb0 = b2[0], bb1 = b2[1];
#pragma unroll
            for (int r = 0; r < 8; ++r) {
                const int M    = r + hi * 8;
                const int eidx = tile * 16 + M;
                float l0 = plog[r].x + bb0;
                float l1 = plog[r].y + bb1;
                float mx = fmaxf(l0, l1);
                float q0 = __expf(l0 - mx), q1 = __expf(l1 - mx);
                float inv = 1.f / (q0 + q1);
                float lp0 = __logf(q0 * inv + 1e-8f);
                float lp1 = __logf(q1 * inv + 1e-8f);
                float g0 = -__logf(-__logf(rng01((unsigned)(eidx * 2 + 0))));
                float g1 = -__logf(-__logf(rng01((unsigned)(eidx * 2 + 1))));
                float z0 = lp0 + g0, z1 = lp1 + g1;     // TEMP = 1.0
                mx = fmaxf(z0, z1);
                q0 = __expf(z0 - mx); q1 = __expf(z1 - mx);
                float p = q0 / (q0 + q1);
                p = fminf(fmaxf(p, 1e-6f), 1.0f);
                poss_out[eidx] = p;
            }
        }
    }
}

// ------------------------ 3) top-K via histogram ---------------------------

__global__ void hist_kernel(const float* __restrict__ poss, unsigned* __restrict__ hist)
{
    int i = blockIdx.x * blockDim.x + threadIdx.x;
    if (i < S_EDGES) {
        int b = min(NBINS - 1, (int)(poss[i] * (float)NBINS));
        atomicAdd(&hist[b], 1u);
    }
}

__global__ void thresh_kernel(const unsigned* __restrict__ hist, int* __restrict__ thr)
{
    if (threadIdx.x == 0 && blockIdx.x == 0) {
        unsigned cum = 0;
        int b = NBINS - 1;
        while (b >= 0) {
            unsigned h = hist[b];
            if (cum + h >= (unsigned)K_INSERT) break;
            cum += h;
            --b;
        }
        if (b < 0) b = 0;
        thr[0] = b;            // boundary bin
        thr[1] = (int)cum;     // count strictly above boundary bin
    }
}

__global__ void select_kernel(const float* __restrict__ poss,
                              const int*   __restrict__ se,
                              const int*   __restrict__ thr,
                              int*         __restrict__ counters,
                              int*         __restrict__ selu,
                              int*         __restrict__ selv,
                              float*       __restrict__ selw)
{
    int i = blockIdx.x * blockDim.x + threadIdx.x;
    if (i >= S_EDGES) return;
    float p = poss[i];
    int b = min(NBINS - 1, (int)(p * (float)NBINS));
    const int bstar = thr[0];
    const int above = thr[1];
    int idx = -1;
    if (b > bstar) {
        idx = atomicAdd(&counters[0], 1);
    } else if (b == bstar) {
        int j = atomicAdd(&counters[1], 1);
        if (j < K_INSERT - above) idx = above + j;
    }
    if (idx >= 0 && idx < K_INSERT) {
        selu[idx] = se[i];
        selv[idx] = se[S_EDGES + i];
        selw[idx] = p;
    }
}

// ----------------- 4) hash coalesce(mean) of undirected edges --------------

__global__ void und_insert_kernel(const int*   __restrict__ selu,
                                  const int*   __restrict__ selv,
                                  const float* __restrict__ selw,
                                  unsigned long long* __restrict__ keys,
                                  float* __restrict__ sums,
                                  int*   __restrict__ hcnt)
{
    int i = blockIdx.x * blockDim.x + threadIdx.x;
    if (i >= 2 * K_INSERT) return;
    int a, b; float w;
    if (i < K_INSERT) { a = selu[i]; b = selv[i]; w = selw[i]; }
    else              { a = selv[i - K_INSERT]; b = selu[i - K_INSERT]; w = selw[i - K_INSERT]; }
    unsigned long long key = (unsigned long long)a * (unsigned long long)N_NODES
                           + (unsigned long long)b + 1ull;      // non-zero
    unsigned h = (unsigned)((key * 0x9E3779B97F4A7C15ull) >> 40) & (HASH_CAP - 1);
    for (;;) {
        unsigned long long prev = atomicCAS(&keys[h], 0ull, key);
        if (prev == 0ull || prev == key) {
            atomicAdd(&sums[h], w);
            atomicAdd(&hcnt[h], 1);
            break;
        }
        h = (h + 1) & (HASH_CAP - 1);
    }
}

__global__ void und_emit_kernel(const unsigned long long* __restrict__ keys,
                                const float* __restrict__ sums,
                                const int*   __restrict__ hcnt,
                                int*   __restrict__ counter,
                                int*   __restrict__ ei_out,
                                float* __restrict__ ew_out)
{
    int h = blockIdx.x * blockDim.x + threadIdx.x;
    if (h >= HASH_CAP) return;
    unsigned long long key = keys[h];
    if (key == 0ull) return;
    int idx = atomicAdd(counter, 1);
    if (idx >= 2 * K_INSERT) return;
    key -= 1ull;
    int a = (int)(key / (unsigned long long)N_NODES);
    int b = (int)(key % (unsigned long long)N_NODES);
    int j = E_EDGES + idx;
    ei_out[j]        = a;
    ei_out[ETOT + j] = b;
    ew_out[j] = sums[h] / (float)max(hcnt[h], 1);
}

// ------------------------------ 5) emit outputs ----------------------------

__global__ void copy_x_kernel(const float4* __restrict__ src, float4* __restrict__ dst)
{
    int i = blockIdx.x * blockDim.x + threadIdx.x;
    if (i < (N_NODES * D_X) / 4) dst[i] = src[i];
}

__global__ void emit_base_kernel(const int*   __restrict__ edge_index,
                                 const float* __restrict__ ew_in,
                                 int*   __restrict__ ei_out,
                                 float* __restrict__ ew_out)
{
    int j = blockIdx.x * blockDim.x + threadIdx.x;
    if (j >= ETOT) return;
    if (j < E_EDGES) {
        ei_out[j]        = edge_index[j];
        ei_out[ETOT + j] = edge_index[E_EDGES + j];
        ew_out[j]        = ew_in[j];
    } else {
        ei_out[j]        = 0;
        ei_out[ETOT + j] = 0;
        ew_out[j]        = 0.f;
    }
}

// ------------------------------- launcher ----------------------------------

extern "C" void kernel_launch(void* const* d_in, const int* in_sizes, int n_in,
                              void* d_out, int out_size, void* d_ws, size_t ws_size,
                              hipStream_t stream)
{
    (void)in_sizes; (void)n_in; (void)out_size; (void)ws_size;

    const float* x   = (const float*)d_in[0];
    const float* eig = (const float*)d_in[1];
    const float* ewi = (const float*)d_in[2];
    const float* W1  = (const float*)d_in[3];
    const float* b1  = (const float*)d_in[4];
    const float* W2  = (const float*)d_in[5];
    const float* b2  = (const float*)d_in[6];
    const int*   ei  = (const int*)d_in[7];
    const int*   se  = (const int*)d_in[8];

    // workspace layout (16-byte aligned chunks)
    char* ws = (char*)d_ws;
    size_t off = 0;
    _Float16* w1h  = (_Float16*)(ws + off);          off += (size_t)IN_DIM * H_DIM * 2; // 160KB
    float*    poss = (float*)(ws + off);             off += (size_t)S_EDGES * 4;
    const size_t zero_begin = off;
    unsigned* hist = (unsigned*)(ws + off);          off += NBINS * 4;
    int*      thr  = (int*)(ws + off);               off += 16;
    int*      cnts = (int*)(ws + off);               off += 16;
    int*      selu = (int*)(ws + off);               off += (size_t)K_INSERT * 4;
    int*      selv = (int*)(ws + off);               off += (size_t)K_INSERT * 4;
    float*    selw = (float*)(ws + off);             off += (size_t)K_INSERT * 4;
    unsigned long long* hkeys = (unsigned long long*)(ws + off); off += (size_t)HASH_CAP * 8;
    float*    hsums = (float*)(ws + off);            off += (size_t)HASH_CAP * 4;
    int*      hcnt  = (int*)(ws + off);              off += (size_t)HASH_CAP * 4;
    const size_t zero_bytes = off - zero_begin;

    // output layout: x | ei(2 x ETOT, int bits) | ew(ETOT)
    float* out_x  = (float*)d_out;
    int*   out_ei = (int*)((float*)d_out + (size_t)N_NODES * D_X);
    float* out_ew = (float*)d_out + (size_t)N_NODES * D_X + 2 * (size_t)ETOT;

    // zero histogram / counters / hash table (graph-capture safe)
    hipMemsetAsync(ws + zero_begin, 0, zero_bytes, stream);

    // 1) W1 -> f16 column-major (one small kernel; stream-ordered before MLP)
    cvt_w1_kernel<<<(IN_DIM * H_DIM + 255) / 256, 256, 0, stream>>>(W1, w1h);

    // 2) persistent fused gather + WMMA MLP -> poss
    {
        const size_t shmem = (size_t)IN_DIM * H_DIM * sizeof(_Float16);  // 160KB
        mlp_wmma_kernel<<<MLP_BLOCKS, 128, shmem, stream>>>(x, eig, w1h, b1, W2, b2,
                                                            se, poss);
    }

    // 3) top-K selection
    hist_kernel<<<(S_EDGES + 255) / 256, 256, 0, stream>>>(poss, hist);
    thresh_kernel<<<1, 32, 0, stream>>>(hist, thr);
    select_kernel<<<(S_EDGES + 255) / 256, 256, 0, stream>>>(poss, se, thr, cnts,
                                                             selu, selv, selw);

    // 5a) base outputs
    copy_x_kernel<<<((N_NODES * D_X / 4) + 255) / 256, 256, 0, stream>>>(
        (const float4*)x, (float4*)out_x);
    emit_base_kernel<<<(ETOT + 255) / 256, 256, 0, stream>>>(ei, ewi, out_ei, out_ew);

    // 4) coalesce(mean) of undirected inserted edges, then append
    und_insert_kernel<<<(2 * K_INSERT + 255) / 256, 256, 0, stream>>>(
        selu, selv, selw, hkeys, hsums, hcnt);
    und_emit_kernel<<<(HASH_CAP + 255) / 256, 256, 0, stream>>>(
        hkeys, hsums, hcnt, &cnts[2], out_ei, out_ew);
}
